// Attention_6141803233961
// MI455X (gfx1250) — matrix operand order repaired and checked by reading, not compiled, not run on hardware
//
#include <hip/hip_runtime.h>

// ---------------------------------------------------------------------------
// Types
// ---------------------------------------------------------------------------
typedef unsigned short u16t;
typedef unsigned int   u32t;
typedef unsigned long long u64t;
typedef __attribute__((ext_vector_type(16))) __bf16 bf16x16;
typedef __attribute__((ext_vector_type(8)))  float  f32x8;
typedef __attribute__((ext_vector_type(8)))  u32t   u32x8;
typedef __attribute__((ext_vector_type(4)))  u32t   u32x4;
typedef __attribute__((ext_vector_type(8)))  int    i32x8;
typedef __attribute__((ext_vector_type(4)))  int    i32x4;

#define BB   4
#define NN   4096
#define DIN  1024
#define DINN 512
#define DOUT 1024
#define MTOT (BB * NN)          // 16384
#define QKVW (3 * DINN)         // 1536

#if __has_builtin(__builtin_amdgcn_tensor_load_to_lds)
#define HAS_TDM 1
#else
#define HAS_TDM 0
#endif

// ---------------------------------------------------------------------------
// Helpers
// ---------------------------------------------------------------------------
__device__ __forceinline__ u16t f2bf(float f) {
    u32t u = __builtin_bit_cast(u32t, f);
    u32t r = u + 0x7FFFu + ((u >> 16) & 1u);   // round-to-nearest-even
    return (u16t)(r >> 16);
}

__device__ __forceinline__ f32x8 zero8() {
    f32x8 c;
#pragma unroll
    for (int v = 0; v < 8; ++v) c[v] = 0.0f;
    return c;
}

// CDNA5 direct global->LDS async copy (ASYNCcnt); 16 bytes per lane.
__device__ __forceinline__ void async_ld_b128(const u16t* gptr, u16t* lptr) {
    u32t loff = (u32t)(uintptr_t)lptr;
    u64t ga   = (u64t)(uintptr_t)gptr;
    asm volatile("global_load_async_to_lds_b128 %0, %1, off"
                 :: "v"(loff), "v"(ga) : "memory");
}
__device__ __forceinline__ void wait_async0() {
    asm volatile("s_wait_asynccnt 0x0" ::: "memory");
}
__device__ __forceinline__ void wait_tensor0() {
#if __has_builtin(__builtin_amdgcn_s_wait_tensorcnt)
    __builtin_amdgcn_s_wait_tensorcnt(0);
#else
    asm volatile("s_wait_tensorcnt 0x0" ::: "memory");
#endif
}

#if HAS_TDM
// TDM 2D tile load: rows x 512 bf16 tile out of the [N x 1536] qkv tensor,
// LDS rows padded to 520 ushorts via the D# pad engine
// (pad_interval=7 -> every 256 DWORDs, pad_amount=3 -> +4 DWORDs = 16B).
// This toolchain exposes the 6-arg builtin:
//   (uint32x4 g0, int32x8 g1, int32x4 g2, int32x4 g3, int32x8, int cpol)
__device__ __forceinline__ void tdm_load_tile(u32t lds_off, const u16t* gptr,
                                              u32t tile_rows) {
    u64t ga = (u64t)(uintptr_t)gptr;
    u32x4 g0;
    g0[0] = 1u;                                 // count=1, user descriptor
    g0[1] = lds_off;                            // lds_addr (bytes)
    g0[2] = (u32t)ga;                           // global_addr[31:0]
    g0[3] = (u32t)(ga >> 32) | 0x80000000u;     // global_addr[56:32] | type=2
    i32x8 g1;
    g1[0] = (int)0x07D10000u;   // data_size=2B | pad_enable | interval=7 | amount=3
    g1[1] = (int)(512u << 16);  // tensor_dim0 = 512 (bits 79:48, low half)
    g1[2] = (int)(4096u << 16); // tensor_dim0 hi | tensor_dim1 = 4096 (low half)
    g1[3] = (int)(512u << 16);  // tensor_dim1 hi | tile_dim0 = 512
    g1[4] = (int)tile_rows;     // tile_dim1 = rows; tile_dim2 = 0
    g1[5] = (int)QKVW;          // tensor_dim0_stride = 1536 (bits 207:160, low)
    g1[6] = 0;
    g1[7] = 0;
    i32x4 z4 = {0, 0, 0, 0};
    i32x8 z8 = {0, 0, 0, 0, 0, 0, 0, 0};
    __builtin_amdgcn_tensor_load_to_lds(g0, g1, z4, z4, z8, 0);
}
#endif

// A-fragment, 16x32 bf16 (ISA 7.12.2): lane = M (0..15), lane[4] picks K-half.
__device__ __forceinline__ bf16x16 frag_A(const u16t* p, int stride, int lane) {
    int m = lane & 15, hi = lane >> 4;
    const u16t* r = p + m * stride + hi * 8;
    u32x8 u;
#pragma unroll
    for (int j = 0; j < 8; ++j) {
        int off = 2 * j + ((j >= 4) ? 8 : 0);
        u[j] = *(const u32t*)(r + off);
    }
    return __builtin_bit_cast(bf16x16, u);
}

// B-fragment, 32x16 bf16: lane&15 = N column, lane[4] picks K-half (same K order as A).
__device__ __forceinline__ bf16x16 frag_B(const u16t* p, int stride, int lane) {
    int n = lane & 15, hi = lane >> 4;
    const u16t* r = p + n * stride + hi * 8;
    u32x8 u;
#pragma unroll
    for (int j = 0; j < 8; ++j) u[j] = *(const u32t*)(r + 2 * j + ((j >= 4) ? 8 : 0));
    return __builtin_bit_cast(bf16x16, u);
}

__device__ __forceinline__ f32x8 wmma_bf16(bf16x16 a, bf16x16 b, f32x8 c) {
    return __builtin_amdgcn_wmma_f32_16x16x32_bf16(
        false, a, false, b, (short)0, c, false, false);
}

// ---------------------------------------------------------------------------
// Kernel 1: fp32 -> bf16 conversion
// ---------------------------------------------------------------------------
__global__ void cvt_f32_bf16(const float* __restrict__ in, u16t* __restrict__ out, int n) {
    int i = blockIdx.x * 256 + threadIdx.x;
    if (i < n) out[i] = f2bf(in[i]);
}

// ---------------------------------------------------------------------------
// Kernel 2/4: bf16 GEMM  C[M,Nn] = A[M,K] * B[K,Nn] (+bias)
// ---------------------------------------------------------------------------
__global__ __launch_bounds__(128) void gemm_bf16(
    const u16t* __restrict__ A, const u16t* __restrict__ B,
    float* __restrict__ outF, u16t* __restrict__ outB,
    const float* __restrict__ bias, int M, int Nn, int K, int outIsF32)
{
    __shared__ u16t As[64 * 40];   // [row][k], stride 40
    __shared__ u16t Bt[64 * 40];   // [n][k] transposed, stride 40

    const int tid  = threadIdx.x;
    const int lane = tid & 31;
    const int w    = tid >> 5;
    const int m0   = blockIdx.y * 64;
    const int n0   = blockIdx.x * 64;
    const int wm   = (w & 1) * 32;
    const int wn   = (w >> 1) * 32;

    f32x8 acc[2][2];
#pragma unroll
    for (int i = 0; i < 2; ++i)
#pragma unroll
        for (int j = 0; j < 2; ++j) acc[i][j] = zero8();

    for (int k0 = 0; k0 < K; k0 += 32) {
        // ---- stage A tile: async direct-to-LDS ----
#pragma unroll
        for (int it = 0; it < 2; ++it) {
            int idx = tid + it * 128;
            int r = idx >> 2, c = idx & 3;
            async_ld_b128(A + (size_t)(m0 + r) * K + k0 + c * 8,
                          As + r * 40 + c * 8);
        }
        // ---- stage B tile transposed: batch both loads before scatter ----
        {
            uint4 d[2];
            int rr[2], cc[2];
#pragma unroll
            for (int it = 0; it < 2; ++it) {
                int idx = tid + it * 128;
                rr[it] = idx >> 3; cc[it] = idx & 7;
                d[it] = *(const uint4*)(B + (size_t)(k0 + rr[it]) * Nn + n0 + cc[it] * 8);
            }
#pragma unroll
            for (int it = 0; it < 2; ++it) {
                const u16t* pv = (const u16t*)&d[it];
#pragma unroll
                for (int e = 0; e < 8; ++e) Bt[(cc[it] * 8 + e) * 40 + rr[it]] = pv[e];
            }
        }
        if (k0 + 32 < K) {
            __builtin_prefetch(A + (size_t)(m0 + (tid >> 1)) * K + k0 + 32, 0, 0);
        }
        wait_async0();
        __syncthreads();

        bf16x16 a0 = frag_A(As + wm * 40, 40, lane);
        bf16x16 a1 = frag_A(As + (wm + 16) * 40, 40, lane);
        bf16x16 b0 = frag_B(Bt + wn * 40, 40, lane);
        bf16x16 b1 = frag_B(Bt + (wn + 16) * 40, 40, lane);
        acc[0][0] = wmma_bf16(a0, b0, acc[0][0]);
        acc[0][1] = wmma_bf16(a0, b1, acc[0][1]);
        acc[1][0] = wmma_bf16(a1, b0, acc[1][0]);
        acc[1][1] = wmma_bf16(a1, b1, acc[1][1]);
        __syncthreads();
    }

    const int hi = lane >> 4, nl = lane & 15;
#pragma unroll
    for (int i = 0; i < 2; ++i)
#pragma unroll
        for (int j = 0; j < 2; ++j) {
            int col = n0 + wn + j * 16 + nl;
#pragma unroll
            for (int v = 0; v < 8; ++v) {
                int row = m0 + wm + i * 16 + v + hi * 8;
                if (outIsF32) {
                    float bv = bias ? bias[col] : 0.0f;
                    outF[(size_t)row * Nn + col] = acc[i][j][v] + bv;
                } else {
                    outB[(size_t)row * Nn + col] = f2bf(acc[i][j][v]);
                }
            }
        }
}

// ---------------------------------------------------------------------------
// Kernel 3: causal flash attention, d=512.  Q/K tiles via TDM tensor_load_to_lds
// (pad engine reproduces the 520-ushort LDS row stride); V transposed by VALU.
// O accumulator resident in LDS fp32 -- 288 KB LDS per workgroup (CDNA5 WGP).
// ---------------------------------------------------------------------------
#define QS 520   // Q/K LDS row stride (ushorts)
#define OS 516   // O LDS row stride (floats)
#define VS 40    // Vt / Ps column stride (ushorts)
#define SS 36    // S scratch row stride (floats)
#define ATTN_LDS_BYTES 288000

__global__ __launch_bounds__(128) void flash_attn(
    const u16t* __restrict__ qkv, u16t* __restrict__ o)
{
    extern __shared__ char smem[];
    u16t*  Qs      = (u16t*)smem;                       // 64 x QS
    u16t*  Ks      = Qs + 64 * QS;                      // 32 x QS
    u16t*  Vt      = Ks + 32 * QS;                      // 512 x VS (transposed)
    float* Os      = (float*)(Vt + 512 * VS);           // 64 x OS
    float* Ssm     = Os + 64 * OS;                      // 64 x SS
    u16t*  Ps      = (u16t*)(Ssm + 64 * SS);            // 64 x VS
    float* mstate  = (float*)(Ps + 64 * VS);            // 64
    float* lstate  = mstate + 64;                       // 64
    float* alpha_s = lstate + 64;                       // 64

    const int tid  = threadIdx.x;
    const int lane = tid & 31;
    const int rw   = (tid >> 5) * 16;
    const int b    = blockIdx.y;
    const int m0   = blockIdx.x * 64;
    const u16t* qkvB = qkv + (size_t)b * NN * QKVW;
    const float scale = 0.04419417382415922f;   // 1/sqrt(512)

    // ---- load Q block (64 x 512 bf16) ----
#if HAS_TDM
    if (tid < 32) {
        tdm_load_tile((u32t)(uintptr_t)Qs, qkvB + (size_t)m0 * QKVW, 64u);
    }
#else
#pragma unroll
    for (int it = 0; it < 32; ++it) {
        int idx = tid + it * 128;
        int r = idx >> 6, c = idx & 63;
        async_ld_b128(qkvB + (size_t)(m0 + r) * QKVW + c * 8, Qs + r * QS + c * 8);
    }
#endif
    // ---- init O accumulator + stats ----
    for (int idx = tid; idx < 64 * OS; idx += 128) Os[idx] = 0.0f;
    if (tid < 64) { mstate[tid] = -3.0e38f; lstate[tid] = 0.0f; }
#if HAS_TDM
    if (tid < 32) wait_tensor0();
#else
    wait_async0();
#endif
    __syncthreads();

    const int jend = m0 + 64;                 // causal: keys <= m0+63
    for (int j0 = 0; j0 < jend; j0 += 32) {
        // ---- K tile via TDM; V tile transposed (batched scatter) ----
#if HAS_TDM
        if (tid < 32) {
            tdm_load_tile((u32t)(uintptr_t)Ks, qkvB + (size_t)j0 * QKVW + DINN, 32u);
        }
#else
#pragma unroll
        for (int it = 0; it < 16; ++it) {
            int idx = tid + it * 128;
            int r = idx >> 6, c = idx & 63;
            async_ld_b128(qkvB + (size_t)(j0 + r) * QKVW + DINN + c * 8,
                          Ks + r * QS + c * 8);
        }
#endif
#pragma unroll
        for (int g = 0; g < 4; ++g) {
            uint4 d[4];
            int rr[4], cc[4];
#pragma unroll
            for (int q = 0; q < 4; ++q) {
                int idx = tid + (g * 4 + q) * 128;
                rr[q] = idx >> 6; cc[q] = idx & 63;
                d[q] = *(const uint4*)(qkvB + (size_t)(j0 + rr[q]) * QKVW + 2 * DINN + cc[q] * 8);
            }
#pragma unroll
            for (int q = 0; q < 4; ++q) {
                const u16t* pv = (const u16t*)&d[q];
#pragma unroll
                for (int e = 0; e < 8; ++e) Vt[(cc[q] * 8 + e) * VS + rr[q]] = pv[e];
            }
        }
#if HAS_TDM
        if (tid < 32) wait_tensor0();
#else
        wait_async0();
#endif
        __syncthreads();

        // ---- S = Q * K^T  (16 rows x 32 keys per wave) ----
        f32x8 s0 = zero8(), s1 = zero8();
#pragma unroll
        for (int kb = 0; kb < 16; ++kb) {
            bf16x16 aq  = frag_A(Qs + rw * QS + kb * 32, QS, lane);
            bf16x16 bk0 = frag_B(Ks + kb * 32, QS, lane);
            bf16x16 bk1 = frag_B(Ks + 16 * QS + kb * 32, QS, lane);
            s0 = wmma_bf16(aq, bk0, s0);
            s1 = wmma_bf16(aq, bk1, s1);
        }

        // ---- scale + causal mask -> LDS S scratch ----
        {
            int nl = lane & 15, hi = lane >> 4;
#pragma unroll
            for (int v = 0; v < 8; ++v) {
                int lr = rw + v + hi * 8;
                int gr = m0 + lr;
                float x0 = s0[v] * scale;
                if (j0 + nl > gr) x0 = -3.0e38f;
                Ssm[lr * SS + nl] = x0;
                float x1 = s1[v] * scale;
                if (j0 + 16 + nl > gr) x1 = -3.0e38f;
                Ssm[lr * SS + 16 + nl] = x1;
            }
        }
        __syncthreads();

        // ---- online softmax (one lane per row) ----
        if ((tid & 31) < 16) {
            int lr = rw + (tid & 15);
            float mold = mstate[lr], lold = lstate[lr];
            float mt = -3.0e38f;
#pragma unroll
            for (int c = 0; c < 32; ++c) mt = fmaxf(mt, Ssm[lr * SS + c]);
            float mn = fmaxf(mold, mt);
            float al = __expf(mold - mn);
            float sum = 0.0f;
#pragma unroll
            for (int c = 0; c < 32; ++c) {
                float p = __expf(Ssm[lr * SS + c] - mn);
                sum += p;
                Ps[lr * VS + c] = f2bf(p);
            }
            mstate[lr] = mn;
            lstate[lr] = al * lold + sum;
            alpha_s[lr] = al;
        }
        __syncthreads();

        // ---- O = O*alpha + P * V  (alpha folded into WMMA C operand) ----
        {
            bf16x16 ap = frag_A(Ps + rw * VS, VS, lane);
            int hi = lane >> 4, nl = lane & 15;
            float a8[8];
#pragma unroll
            for (int v = 0; v < 8; ++v) a8[v] = alpha_s[rw + v + hi * 8];
#pragma unroll 4
            for (int nt = 0; nt < 32; ++nt) {
                int n0c = nt * 16;
                bf16x16 bv = frag_B(Vt + n0c * VS, VS, lane);
                float* cp = Os + rw * OS + n0c + nl;
                f32x8 c;
#pragma unroll
                for (int v = 0; v < 8; ++v) c[v] = cp[(v + hi * 8) * OS] * a8[v];
                c = wmma_bf16(ap, bv, c);
#pragma unroll
                for (int v = 0; v < 8; ++v) cp[(v + hi * 8) * OS] = c[v];
            }
        }
        __syncthreads();
    }

    // ---- normalize by 1/l and emit bf16 O block ----
    for (int it = 0; it < 256; ++it) {
        int idx = tid + it * 128;
        int r = idx >> 9, c = idx & 511;
        float ov = Os[r * OS + c] / lstate[r];
        o[((size_t)b * NN + m0 + r) * DINN + c] = f2bf(ov);
    }
}

// ---------------------------------------------------------------------------
// Launch: cvt -> QKV GEMM -> flash attention -> out GEMM (+bias, fp32)
// ---------------------------------------------------------------------------
extern "C" void kernel_launch(void* const* d_in, const int* in_sizes, int n_in,
                              void* d_out, int out_size, void* d_ws, size_t ws_size,
                              hipStream_t stream) {
    (void)in_sizes; (void)n_in; (void)out_size; (void)ws_size;
    const float* x     = (const float*)d_in[0];
    const float* w_qkv = (const float*)d_in[1];
    const float* w_out = (const float*)d_in[2];
    const float* b_out = (const float*)d_in[3];
    float* out = (float*)d_out;

    const size_t Sx  = (size_t)MTOT * DIN;
    const size_t Swq = (size_t)DIN * QKVW;
    const size_t Swo = (size_t)DINN * DOUT;
    const size_t Sqk = (size_t)MTOT * QKVW;
    u16t* xb   = (u16t*)d_ws;
    u16t* wqb  = xb + Sx;
    u16t* wob  = wqb + Swq;
    u16t* qkvb = wob + Swo;
    u16t* ob   = qkvb + Sqk;

    cvt_f32_bf16<<<(int)((Sx  + 255) / 256), 256, 0, stream>>>(x,     xb,  (int)Sx);
    cvt_f32_bf16<<<(int)((Swq + 255) / 256), 256, 0, stream>>>(w_qkv, wqb, (int)Swq);
    cvt_f32_bf16<<<(int)((Swo + 255) / 256), 256, 0, stream>>>(w_out, wob, (int)Swo);

    // QKV projection: [16384,1024] x [1024,1536] -> bf16
    gemm_bf16<<<dim3(QKVW / 64, MTOT / 64), 128, 0, stream>>>(
        xb, wqb, nullptr, qkvb, nullptr, MTOT, QKVW, DIN, 0);

    // Flash attention: grid (N/64, B), 288 KB dynamic LDS
    flash_attn<<<dim3(NN / 64, BB), 128, ATTN_LDS_BYTES, stream>>>(qkvb, ob);

    // Output projection + bias: [16384,512] x [512,1024] -> fp32 d_out
    gemm_bf16<<<dim3(DOUT / 64, MTOT / 64), 128, 0, stream>>>(
        ob, wob, out, nullptr, b_out, MTOT, DOUT, DINN, 1);
}
